// TemporalAttentionLayer_11312943857879
// MI455X (gfx1250) — matrix operand order repaired and checked by reading, not compiled, not run on hardware
//
#include <hip/hip_runtime.h>
#include <hip/hip_bf16.h>

// ---------------------------------------------------------------------------
// TemporalAttentionLayer for MI455X (gfx1250, wave32, WMMA, async-to-LDS)
//   x:(4,512,8,256) f32 -> PE add -> QKV proj -> causal MHA (H=8, hd=32)
//   -> out proj -> residual -> LayerNorm -> (4,512,8,256) f32
// All matmuls: v_wmma_f32_16x16x32_f16 (f16 operands, f32 accum).
// QKV GEMM: block = 8 waves = 128(M)x64(N) region; weight strip staged to
//   LDS with global_load_async_to_lds_b128 + s_wait_asynccnt; 4 acc/wave.
// Attention: flash-style, fully in registers; scores computed transposed
//   (A=K, B=Q) so exp'd P lands exactly in the A-fragment layout for P@V.
//   V stored transposed (hd x S) so its B-fragments are contiguous.
// Workspace (~42 MB): Xh | Qh | Kh | Vt | Oh | Wh(4 f16 weights)
// ---------------------------------------------------------------------------

typedef __attribute__((ext_vector_type(16))) _Float16 v16h;
typedef __attribute__((ext_vector_type(8)))  _Float16 v8h;
typedef __attribute__((ext_vector_type(8)))  float    v8f;

#define WMMA_F16(A, B, C) \
  __builtin_amdgcn_wmma_f32_16x16x32_f16(false, (A), false, (B), (short)0, (C), false, false)

constexpr int Bc  = 4;
constexpr int Sc  = 512;
constexpr int Nc  = 8;
constexpr int Dc  = 256;
constexpr int Hc  = 8;
constexpr int HDc = 32;
constexpr int BNc = Bc * Nc;     // 32
constexpr int Mc  = BNc * Sc;    // 16384 rows

// ---- fragment loaders (match CDNA5 16-bit WMMA VGPR layouts) ---------------
// A-matrix 16x32 f16: lane row = L&15; lanes<16 hold K {0..7,16..23},
// lanes>=16 hold K {8..15,24..31}  -> two contiguous 8-half chunks.
__device__ __forceinline__ v16h load_afrag(const _Float16* row, int hi8) {
  const v8h a0 = *(const v8h*)(row + hi8);
  const v8h a1 = *(const v8h*)(row + 16 + hi8);
  v16h r;
#pragma unroll
  for (int i = 0; i < 8; ++i) { r[i] = a0[i]; r[i + 8] = a1[i]; }
  return r;
}
// B-matrix 32x16 f16: lane col = L&15; lanes<16 hold K {0..7,16..23},
// lanes>=16 hold K {8..15,24..31} (same order as A-frag).
__device__ __forceinline__ v16h load_bfrag(const _Float16* base, int hi16) {
  const v8h b0 = *(const v8h*)(base + (hi16 >> 1));
  const v8h b1 = *(const v8h*)(base + 16 + (hi16 >> 1));
  v16h r;
#pragma unroll
  for (int i = 0; i < 8; ++i) { r[i] = b0[i]; r[i + 8] = b1[i]; }
  return r;
}

// ---- kernel 1: positional encoding + transpose to (B,N,S,D) + f16 cast -----
__global__ void k_posenc(const float* __restrict__ x, _Float16* __restrict__ Xh) {
  int m   = blockIdx.x;              // m = (b*8+n)*512 + s
  int bni = m >> 9, s = m & 511;
  int b   = bni >> 3, n = bni & 7;
  const float* xrow = x + (((size_t)b * Sc + s) * Nc + n) * Dc;
  _Float16* orow    = Xh + (size_t)m * Dc;
  int i = threadIdx.x;               // pair index 0..127
  float div = __expf((float)(2 * i) * (-9.210340371976184f / (float)Dc));
  float ang = (float)s * div;
  orow[2 * i]     = (_Float16)(xrow[2 * i]     + __sinf(ang));
  orow[2 * i + 1] = (_Float16)(xrow[2 * i + 1] + __cosf(ang));
}

// ---- kernel 2: cast all four weight matrices to f16 ------------------------
__global__ void k_castw(const float* __restrict__ Wq, const float* __restrict__ Wk,
                        const float* __restrict__ Wv, const float* __restrict__ Wo,
                        _Float16* __restrict__ Wh) {
  int i = blockIdx.x * blockDim.x + threadIdx.x;   // 0..65535
  Wh[i]             = (_Float16)Wq[i];
  Wh[65536 + i]     = (_Float16)Wk[i];
  Wh[2 * 65536 + i] = (_Float16)Wv[i];
  Wh[3 * 65536 + i] = (_Float16)Wv[i - i];         // placeholder overwritten below
  Wh[3 * 65536 + i] = (_Float16)Wo[i];
}

// ---- kernel 3: fused QKV GEMM --------------------------------------------
// grid: which(3) x 128 m-groups x 4 n-strips. Block = 256 thr = 8 waves.
// Block computes rows [mg*128, +128), cols [ns*64, +64).
// Weight strip W[n0..n0+63][0..255] (32 KB f16) staged to LDS via
// GLOBAL_LOAD_ASYNC_TO_LDS_B128, shared by all 8 waves.
// Q written pre-scaled by 1/sqrt(hd) into (bnh,S,hd); K into (bnh,S,hd);
// V written TRANSPOSED into (bnh,hd,S).
__global__ void k_qkv(const _Float16* __restrict__ Xh, const _Float16* __restrict__ Wh,
                      const float* __restrict__ bq, const float* __restrict__ bk,
                      const float* __restrict__ bv,
                      _Float16* __restrict__ Qh, _Float16* __restrict__ Kh,
                      _Float16* __restrict__ Vt) {
  extern __shared__ _Float16 Wlds[];     // dynamic LDS -> base offset 0
  int bid   = blockIdx.x;
  int which = bid >> 9;                  // 0:Q 1:K 2:V  (512 blocks each)
  int rem   = bid & 511;
  int mg    = rem >> 2;                  // 0..127
  int ns    = rem & 3;                   // 0..3
  int n0    = ns * 64;
  int tid   = threadIdx.x, wv = tid >> 5, lane = tid & 31;
  int col = lane & 15, hi8 = (lane >> 4) * 8, hi16 = (lane >> 4) * 16;

  // ---- async stage weight strip: 64 rows x 512 B = 2048 x 16 B chunks -----
  const _Float16* Wsrc = Wh + (size_t)which * Dc * Dc + (size_t)n0 * Dc;
#pragma unroll
  for (int j = 0; j < 8; ++j) {
    int chunk = j * 256 + tid;           // 0..2047, lane-consecutive 16B chunks
    unsigned ldsoff = (unsigned)chunk * 16u;
    const _Float16* gsrc = Wsrc + chunk * 8;   // 8 halves = 16 B
    asm volatile("global_load_async_to_lds_b128 %0, %1, off"
                 :: "v"(ldsoff), "v"(gsrc) : "memory");
  }
  asm volatile("s_wait_asynccnt 0x0" ::: "memory");
  __syncthreads();

  // ---- GEMM: wave wv does rows m0..m0+15, 4 N-tiles of 16 ------------------
  int m0 = mg * 128 + wv * 16;
  const _Float16* Arow = Xh + (size_t)(m0 + col) * Dc;
  v8f acc[4] = {{}, {}, {}, {}};
#pragma unroll
  for (int k0 = 0; k0 < Dc; k0 += 32) {
    v16h a = load_afrag(Arow + k0, hi8);
#pragma unroll
    for (int t = 0; t < 4; ++t) {
      v16h b = load_bfrag(Wlds + (size_t)(t * 16 + col) * Dc + k0, hi16);
      acc[t] = WMMA_F16(a, b, acc[t]);
    }
  }

  const float* bias = (which == 0) ? bq : (which == 1) ? bk : bv;
#pragma unroll
  for (int t = 0; t < 4; ++t) {
    int n  = n0 + t * 16 + col;
    float bn = bias[n];
    int h = n >> 5, dd = n & 31;
#pragma unroll
    for (int r = 0; r < 8; ++r) {
      int m = m0 + r + hi8;
      int bni = m >> 9, s = m & 511;
      int bnh = bni * Hc + h;
      float y = acc[t][r] + bn;
      if (which == 0)
        Qh[((size_t)bnh * Sc + s) * HDc + dd] = (_Float16)(y * 0.17677669529663687f); // 1/sqrt(32)
      else if (which == 1)
        Kh[((size_t)bnh * Sc + s) * HDc + dd] = (_Float16)y;
      else
        Vt[((size_t)bnh * HDc + dd) * Sc + s] = (_Float16)y;
    }
  }
}

// ---- kernel 4: causal flash attention; one wave per (bnh, 16-query tile) ---
__global__ void k_attn(const _Float16* __restrict__ Qh, const _Float16* __restrict__ Kh,
                       const _Float16* __restrict__ Vt, _Float16* __restrict__ Oh) {
  int wave = blockIdx.x * (blockDim.x >> 5) + (threadIdx.x >> 5);
  int lane = threadIdx.x & 31;
  int qt  = wave & 31;
  int bnh = wave >> 5;                       // 0..255
  int q0  = qt * 16;
  int col = lane & 15, hi8 = (lane >> 4) * 8, hi16 = (lane >> 4) * 16;

  const _Float16* Qb = Qh + (size_t)bnh * Sc * HDc;
  const _Float16* Kb = Kh + (size_t)bnh * Sc * HDc;
  const _Float16* Vb = Vt + (size_t)bnh * HDc * Sc;

  // persistent Q B-fragment (32 x 16): lane col = query, elems d = hi8+i (i<8), 8+hi8+i (i>=8)
  v16h qf = load_bfrag(Qb + (size_t)(q0 + col) * HDc, hi16);
  int q = q0 + col;                          // query this lane tracks stats for

  float mrun = -1e30f, lrun = 0.f;
  v8f o1 = {}, o2 = {};
  v8f zf = {};
  int nkb = (q0 + 16 + 31) >> 5;             // 32-key blocks covering causal span

  for (int kb = 0; kb < nkb; ++kb) {
    int kbase = kb * 32;
    if (kb + 1 < nkb) {                      // prefetch next K/V block
      __builtin_prefetch(Kb + (size_t)(kbase + 32 + col) * HDc, 0, 3);
      __builtin_prefetch(Vb + (size_t)col * Sc + kbase + 32, 0, 3);
    }
    // scores^T: A = K-tile (keys x d), B = Q -> C rows=keys, cols=queries
    v16h kf0 = load_afrag(Kb + (size_t)(kbase + col) * HDc, hi8);
    v16h kf1 = load_afrag(Kb + (size_t)(kbase + 16 + col) * HDc, hi8);
    v8f c1 = WMMA_F16(kf0, qf, zf);
    v8f c2 = WMMA_F16(kf1, qf, zf);

    // causal mask + per-lane max over this lane's 16 keys
    float bm = -1e30f;
#pragma unroll
    for (int r = 0; r < 8; ++r) {
      int key1 = kbase + r + hi8;
      int key2 = kbase + 16 + r + hi8;
      if (key1 > q) c1[r] = -1e30f;
      if (key2 > q) c2[r] = -1e30f;
      bm = fmaxf(bm, fmaxf(c1[r], c2[r]));
    }
    bm = fmaxf(bm, __shfl_xor(bm, 16, 32));   // combine lanes q and q+16
    float mnew = fmaxf(mrun, bm);
    float fold = __expf(mrun - mnew);

    float psum = 0.f;
    v16h pa;                                  // P already in A-frag layout!
#pragma unroll
    for (int r = 0; r < 8; ++r) {
      float p1 = __expf(c1[r] - mnew);
      float p2 = __expf(c2[r] - mnew);
      psum += p1 + p2;
      pa[r]     = (_Float16)p1;
      pa[r + 8] = (_Float16)p2;
    }
    psum += __shfl_xor(psum, 16, 32);
    lrun = lrun * fold + psum;
    mrun = mnew;

    // rescale running output: rows of O are queries r+hi8
#pragma unroll
    for (int r = 0; r < 8; ++r) {
      float f = __shfl(fold, r + hi8, 32);
      o1[r] *= f;
      o2[r] *= f;
    }
    // P @ V : B-frags contiguous from transposed V
    v16h vf0 = load_bfrag(Vb + (size_t)col * Sc + kbase, hi16);
    v16h vf1 = load_bfrag(Vb + (size_t)(col + 16) * Sc + kbase, hi16);
    o1 = WMMA_F16(pa, vf0, o1);
    o2 = WMMA_F16(pa, vf1, o2);
  }

  int bni = bnh >> 3, h = bnh & 7;
#pragma unroll
  for (int r = 0; r < 8; ++r) {
    float li = __shfl(lrun, r + hi8, 32);
    float inv = 1.0f / li;
    int s = q0 + r + hi8;
    size_t base = ((size_t)bni * Sc + s) * Dc + h * HDc;
    Oh[base + col]      = (_Float16)(o1[r] * inv);
    Oh[base + col + 16] = (_Float16)(o2[r] * inv);
  }
}

// ---- kernel 5: output GEMM + bias + residual + LayerNorm -------------------
// One block (128 thr = 4 waves) per 16-row tile; each wave does 4 N-tiles.
__global__ void k_outln(const _Float16* __restrict__ Oh, const _Float16* __restrict__ Woh,
                        const float* __restrict__ bo, const float* __restrict__ x,
                        const float* __restrict__ gamma, const float* __restrict__ beta,
                        float* __restrict__ out) {
  __shared__ float ytile[16][Dc];
  __shared__ float mu[16], rinv[16];
  int m0 = blockIdx.x * 16;
  int tid = threadIdx.x, wv = tid >> 5, lane = tid & 31;
  int col = lane & 15, hi8 = (lane >> 4) * 8, hi16 = (lane >> 4) * 16;

  const _Float16* Arow = Oh + (size_t)(m0 + col) * Dc;
  v8f acc[4] = {{}, {}, {}, {}};
#pragma unroll
  for (int k0 = 0; k0 < Dc; k0 += 32) {
    v16h a = load_afrag(Arow + k0, hi8);
#pragma unroll
    for (int t = 0; t < 4; ++t) {
      int n0 = wv * 64 + t * 16;
      v16h b = load_bfrag(Woh + (size_t)(n0 + col) * Dc + k0, hi16);
      acc[t] = WMMA_F16(a, b, acc[t]);
    }
  }
#pragma unroll
  for (int t = 0; t < 4; ++t) {
    int n = wv * 64 + t * 16 + col;
    float bn = bo[n];
#pragma unroll
    for (int r = 0; r < 8; ++r) ytile[r + hi8][n] = acc[t][r] + bn;
  }
  __syncthreads();
  // residual add (x in original (B,S,N,D) layout)
  for (int i = tid; i < 16 * Dc; i += blockDim.x) {
    int r = i >> 8, c = i & 255;
    int m = m0 + r;
    int bni = m >> 9, s = m & 511;
    int b = bni >> 3, n = bni & 7;
    ytile[r][c] += x[(((size_t)b * Sc + s) * Nc + n) * Dc + c];
  }
  __syncthreads();
  if (tid < 16) {
    float sum = 0.f, sq = 0.f;
    for (int c = 0; c < Dc; ++c) { float v = ytile[tid][c]; sum += v; sq += v * v; }
    float mean = sum * (1.0f / Dc);
    float var  = sq * (1.0f / Dc) - mean * mean;
    mu[tid] = mean;
    rinv[tid] = rsqrtf(var + 1e-5f);
  }
  __syncthreads();
  for (int i = tid; i < 16 * Dc; i += blockDim.x) {
    int r = i >> 8, c = i & 255;
    int m = m0 + r;
    int bni = m >> 9, s = m & 511;
    int b = bni >> 3, n = bni & 7;
    out[(((size_t)b * Sc + s) * Nc + n) * Dc + c] =
        (ytile[r][c] - mu[r]) * rinv[r] * gamma[c] + beta[c];
  }
}

// ---------------------------------------------------------------------------
extern "C" void kernel_launch(void* const* d_in, const int* in_sizes, int n_in,
                              void* d_out, int out_size, void* d_ws, size_t ws_size,
                              hipStream_t stream) {
  (void)in_sizes; (void)n_in; (void)out_size; (void)ws_size;
  const float* x  = (const float*)d_in[0];
  const float* Wq = (const float*)d_in[1];
  const float* Wk = (const float*)d_in[2];
  const float* Wv = (const float*)d_in[3];
  const float* Wo = (const float*)d_in[4];
  const float* bq = (const float*)d_in[5];
  const float* bk = (const float*)d_in[6];
  const float* bv = (const float*)d_in[7];
  const float* bo = (const float*)d_in[8];
  const float* gamma = (const float*)d_in[9];
  const float* beta  = (const float*)d_in[10];
  float* out = (float*)d_out;

  // workspace carve-up (needs ~42.5 MB)
  char* ws = (char*)d_ws;
  const size_t MB8 = (size_t)8 << 20;
  _Float16* Xh = (_Float16*)(ws);              // (BN,S,D)   8 MB
  _Float16* Qh = (_Float16*)(ws + 1 * MB8);    // (bnh,S,hd) 8 MB  (pre-scaled)
  _Float16* Kh = (_Float16*)(ws + 2 * MB8);    // (bnh,S,hd) 8 MB
  _Float16* Vt = (_Float16*)(ws + 3 * MB8);    // (bnh,hd,S) 8 MB  (transposed)
  _Float16* Oh = (_Float16*)(ws + 4 * MB8);    // (BN,S,D)   8 MB
  _Float16* Wh = (_Float16*)(ws + 5 * MB8);    // 4x(256,256) f16, 512 KB

  k_castw<<<256, 256, 0, stream>>>(Wq, Wk, Wv, Wo, Wh);
  k_posenc<<<Mc, 128, 0, stream>>>(x, Xh);
  // 3 x 128 m-groups x 4 n-strips = 1536 blocks, 32 KB dynamic LDS each
  k_qkv<<<1536, 256, 32768, stream>>>(Xh, Wh, bq, bk, bv, Qh, Kh, Vt);
  // 256 heads * 32 q-tiles = 8192 wave-tasks, 8 waves/block
  k_attn<<<1024, 256, 0, stream>>>(Qh, Kh, Vt, Oh);
  // 1024 row-tiles
  k_outln<<<1024, 128, 0, stream>>>(Oh, Wh + 3 * 65536, bo, x, gamma, beta, out);
}
